// S6_noC_11699490914652
// MI455X (gfx1250) — matrix operand order repaired and checked by reading, not compiled, not run on hardware
//
#include <hip/hip_runtime.h>
#include <hip/hip_bf16.h>

// Problem constants from the reference
#define BQ 8
#define LQ 2048
#define DQ 768
#define NC 16                 // scan chunks
#define CL (LQ / NC)          // 128 steps per chunk

typedef __attribute__((ext_vector_type(16))) __bf16 v16bf;
typedef __attribute__((ext_vector_type(8)))  float  v8f;

static __device__ __forceinline__ float sigmoidf(float t) {
    return 1.0f / (1.0f + __expf(-t));
}

// ---------------------------------------------------------------------------
// f32 -> bf16 conversion
// ---------------------------------------------------------------------------
__global__ void cvt_bf16_kernel(const float* __restrict__ in,
                                __bf16* __restrict__ out, int n) {
    int i = blockIdx.x * blockDim.x + threadIdx.x;
    if (i < n) out[i] = (__bf16)in[i];
}

// ---------------------------------------------------------------------------
// mean over L: sq[b,d] = (1/L) * sum_t x[b,t,d]
// ---------------------------------------------------------------------------
__global__ void mean_kernel(const float* __restrict__ x,
                            float* __restrict__ sq) {
    int idx = blockIdx.x * blockDim.x + threadIdx.x;   // 0..B*D-1
    if (idx >= BQ * DQ) return;
    int b = idx / DQ, d = idx % DQ;
    const float* p = x + (size_t)b * LQ * DQ + d;
    float s = 0.0f;
    for (int t = 0; t < LQ; ++t) s += p[(size_t)t * DQ];
    sq[idx] = s * (1.0f / (float)LQ);
}

// ---------------------------------------------------------------------------
// SToken gate: g[b,e] = sigmoid(relu(sum_d sq[b,d]*st_W[e,d] + st_b[e])) + st_bias[e]
// ---------------------------------------------------------------------------
__global__ void stoken_kernel(const float* __restrict__ sq,
                              const float* __restrict__ st_W,
                              const float* __restrict__ st_b,
                              const float* __restrict__ st_bias,
                              float* __restrict__ g) {
    int idx = blockIdx.x * blockDim.x + threadIdx.x;   // 0..B*D-1
    if (idx >= BQ * DQ) return;
    int b = idx / DQ, e = idx % DQ;
    const float* sp = sq + (size_t)b * DQ;
    const float* wp = st_W + (size_t)e * DQ;
    float acc = 0.0f;
    for (int d = 0; d < DQ; ++d) acc = fmaf(sp[d], wp[d], acc);
    acc += st_b[e];
    acc = acc > 0.0f ? acc : 0.0f;                     // relu
    g[idx] = sigmoidf(acc) + st_bias[e];
}

// ---------------------------------------------------------------------------
// WMMA bf16 GEMM: Y[m,n] = sum_k Xb[m,k] * Wb[n,k] (+ bias / delta epilogue)
// M = B*L = 16384, N = K = 768.
// Tile: BM=128 x BN=64, BK=32. 256 threads = 8 waves; wave grid 4(M) x 2(N),
// each wave owns a 32x32 patch = 2x2 wmma 16x16 tiles.
// mode 0: Y = acc + bias[n]
// mode 1: Y = sigmoid(acc + bias[n] + dpos[(m % L)*N + n])
// ---------------------------------------------------------------------------
#define BM 128
#define BN 64
#define BK 32
#define LDA (BK + 8)   // padded bf16 row stride (80 B, 16B-aligned, conflict-free)
#define LDB (BK + 8)

__global__ __launch_bounds__(256)
void gemm_bf16_wmma_kernel(const __bf16* __restrict__ Xb,   // [M,K]
                           const __bf16* __restrict__ Wb,   // [N,K]
                           const float* __restrict__ bias,  // [N]
                           const float* __restrict__ dpos,  // [L*N] or nullptr
                           float* __restrict__ Y,           // [M,N]
                           int M, int N, int K, int mode) {
    __shared__ __bf16 As[BM * LDA];
    __shared__ __bf16 Bs[BN * LDB];

    const int tid  = threadIdx.x;
    const int wave = tid >> 5;
    const int lane = tid & 31;
    const int wm   = wave & 3;      // M strip (x32)
    const int wn   = wave >> 2;     // N strip (x32)
    const int m_blk = blockIdx.x * BM;
    const int n_blk = blockIdx.y * BN;

    const int lrow = lane & 15;     // row/col within 16
    const int lhi  = lane >> 4;     // K-half / M-high selector

    v8f acc00 = {}, acc01 = {}, acc10 = {}, acc11 = {};

    union Frag { uint4 u[2]; v16bf v; };

    for (int k0 = 0; k0 < K; k0 += BK) {
        // ---- stage A tile: 128x32 bf16 = 512 x uint4, 2 per thread ----
        #pragma unroll
        for (int it = 0; it < 2; ++it) {
            int q  = tid + it * 256;        // 0..511
            int r  = q >> 2;                // 0..127
            int cq = q & 3;                 // 8-elem chunk
            uint4 v = *reinterpret_cast<const uint4*>(
                Xb + (size_t)(m_blk + r) * K + k0 + cq * 8);
            *reinterpret_cast<uint4*>(&As[r * LDA + cq * 8]) = v;
        }
        // ---- stage B tile: 64x32 bf16 = 256 x uint4, 1 per thread ----
        {
            int r  = tid >> 2;              // 0..63
            int cq = tid & 3;
            uint4 v = *reinterpret_cast<const uint4*>(
                Wb + (size_t)(n_blk + r) * K + k0 + cq * 8);
            *reinterpret_cast<uint4*>(&Bs[r * LDB + cq * 8]) = v;
        }
        __syncthreads();

        // ---- load fragments per CDNA5 16-bit WMMA VGPR layouts ----
        Frag a0, a1, b0, b1;
        {   // A: lane holds row m=lrow; elems 0..7 -> k = lhi*8+0..7,
            // elems 8..15 -> k = 16+lhi*8+0..7 (two contiguous 16B chunks)
            const __bf16* p0 = &As[(wm * 32 +  0 + lrow) * LDA];
            a0.u[0] = *reinterpret_cast<const uint4*>(p0 + lhi * 8);
            a0.u[1] = *reinterpret_cast<const uint4*>(p0 + 16 + lhi * 8);
            const __bf16* p1 = &As[(wm * 32 + 16 + lrow) * LDA];
            a1.u[0] = *reinterpret_cast<const uint4*>(p1 + lhi * 8);
            a1.u[1] = *reinterpret_cast<const uint4*>(p1 + 16 + lhi * 8);
        }
        {   // B: lane holds col n=lrow; elems 0..7 -> k = lhi*8+0..7, elems 8..15 -> k = 16+lhi*8+0..7
            const __bf16* p0 = &Bs[(wn * 32 +  0 + lrow) * LDB + lhi * 8];
            b0.u[0] = *reinterpret_cast<const uint4*>(p0);
            b0.u[1] = *reinterpret_cast<const uint4*>(p0 + 16);
            const __bf16* p1 = &Bs[(wn * 32 + 16 + lrow) * LDB + lhi * 8];
            b1.u[0] = *reinterpret_cast<const uint4*>(p1);
            b1.u[1] = *reinterpret_cast<const uint4*>(p1 + 16);
        }

        acc00 = __builtin_amdgcn_wmma_f32_16x16x32_bf16(
            false, a0.v, false, b0.v, (short)0, acc00, false, false);
        acc01 = __builtin_amdgcn_wmma_f32_16x16x32_bf16(
            false, a0.v, false, b1.v, (short)0, acc01, false, false);
        acc10 = __builtin_amdgcn_wmma_f32_16x16x32_bf16(
            false, a1.v, false, b0.v, (short)0, acc10, false, false);
        acc11 = __builtin_amdgcn_wmma_f32_16x16x32_bf16(
            false, a1.v, false, b1.v, (short)0, acc11, false, false);

        __syncthreads();
    }

    // ---- epilogue: C/D layout lane l, vgpr v -> M = v + 8*(l>>4), N = l&15 ----
    v8f* accs[4] = { &acc00, &acc01, &acc10, &acc11 };
    #pragma unroll
    for (int t = 0; t < 4; ++t) {
        int mt = t >> 1, nt = t & 1;
        int rbase = m_blk + wm * 32 + mt * 16 + 8 * lhi;
        int c     = n_blk + wn * 32 + nt * 16 + lrow;
        float bv = bias[c];
        #pragma unroll
        for (int v = 0; v < 8; ++v) {
            int r = rbase + v;
            float val = (*accs[t])[v] + bv;
            if (mode == 1) {
                int l = r & (LQ - 1);           // r % L (L is power of 2)
                val += dpos[(size_t)l * N + c];
                val = sigmoidf(val);
            }
            Y[(size_t)r * N + c] = val;
        }
    }
}

// ---------------------------------------------------------------------------
// Chunked linear scan. Recurrence s_t = a_t*s + bx_t with a = delta*A,
// bx = delta*B0*x is linear, so each chunk reduces to (P = prod a, Q).
// Phase 1: per (chunk,b,d) compute (P,Q) over CL steps.  NC*B*D threads.
// ---------------------------------------------------------------------------
__global__ void scan_phase1_kernel(const float* __restrict__ x,
                                   const float* __restrict__ delta,
                                   const float* __restrict__ B0v,
                                   const float* __restrict__ Aml,   // [L,D]
                                   float* __restrict__ P,           // [NC, B*D]
                                   float* __restrict__ Q) {         // [NC, B*D]
    int idx = blockIdx.x * blockDim.x + threadIdx.x;   // 0..NC*B*D-1
    if (idx >= NC * BQ * DQ) return;
    int c  = idx / (BQ * DQ);      // chunk (wave-uniform: B*D multiple of 32)
    int bd = idx % (BQ * DQ);
    int b = bd / DQ, d = bd % DQ;
    int t0 = c * CL;
    size_t base = (size_t)b * LQ * DQ + d;
    float p = 1.0f, s = 0.0f;
    for (int t = t0; t < t0 + CL; ++t) {
        size_t i = base + (size_t)t * DQ;
        float de = delta[i];
        float a  = de * Aml[(size_t)t * DQ + d];
        float bx = de * B0v[i] * x[i];
        p *= a;
        s = fmaf(a, s, bx);
    }
    P[idx] = p;
    Q[idx] = s;
}

// ---------------------------------------------------------------------------
// Phase 2: combine (P,Q) prefixes (<= NC-1 steps, wave-uniform trip count),
// replay chunk with correct initial state, write gated output:
// out = (C*s + g*x) * sigmoid(x)
// ---------------------------------------------------------------------------
__global__ void scan_phase2_kernel(const float* __restrict__ x,
                                   const float* __restrict__ delta,
                                   const float* __restrict__ B0v,
                                   const float* __restrict__ Cv,
                                   const float* __restrict__ Aml,   // [L,D]
                                   const float* __restrict__ g,     // [B,D]
                                   const float* __restrict__ P,     // [NC, B*D]
                                   const float* __restrict__ Q,     // [NC, B*D]
                                   float* __restrict__ out) {
    int idx = blockIdx.x * blockDim.x + threadIdx.x;   // 0..NC*B*D-1
    if (idx >= NC * BQ * DQ) return;
    int c  = idx / (BQ * DQ);
    int bd = idx % (BQ * DQ);
    int b = bd / DQ, d = bd % DQ;

    // initial state entering chunk c
    float s = 0.0f;
    for (int j = 0; j < c; ++j) {
        int jj = j * (BQ * DQ) + bd;
        s = fmaf(P[jj], s, Q[jj]);
    }

    float gv = g[bd];
    int t0 = c * CL;
    size_t base = (size_t)b * LQ * DQ + d;
    for (int t = t0; t < t0 + CL; ++t) {
        size_t i  = base + (size_t)t * DQ;
        float xd  = x[i];
        float de  = delta[i];
        float a   = de * Aml[(size_t)t * DQ + d];
        float bx  = de * B0v[i] * xd;
        s = fmaf(a, s, bx);
        float y   = fmaf(Cv[i], s, gv * xd);
        out[i] = y * sigmoidf(xd);
    }
}

// ---------------------------------------------------------------------------
// Host launcher
// ---------------------------------------------------------------------------
extern "C" void kernel_launch(void* const* d_in, const int* in_sizes, int n_in,
                              void* d_out, int out_size, void* d_ws, size_t ws_size,
                              hipStream_t stream) {
    (void)in_sizes; (void)n_in; (void)out_size; (void)ws_size;

    const float* x       = (const float*)d_in[0];   // [B,L,D]
    const float* W_B     = (const float*)d_in[1];   // [D,D]
    const float* b_B     = (const float*)d_in[2];
    const float* W_C     = (const float*)d_in[3];
    const float* b_C     = (const float*)d_in[4];
    const float* W_d     = (const float*)d_in[5];
    const float* b_d     = (const float*)d_in[6];
    const float* delta_p = (const float*)d_in[7];   // [1,L,D]
    const float* A       = (const float*)d_in[8];   // [L,D]
    const float* st_W    = (const float*)d_in[9];
    const float* st_b    = (const float*)d_in[10];
    const float* st_bias = (const float*)d_in[11];
    float* out = (float*)d_out;

    const int M = BQ * LQ;          // 16384
    const int N = DQ, K = DQ;       // 768

    // workspace carve-up (256B aligned)
    char* p = (char*)d_ws;
    auto carve = [&](size_t bytes) -> void* {
        void* r = (void*)p;
        p += (bytes + 255) & ~(size_t)255;
        return r;
    };
    __bf16* xb  = (__bf16*)carve((size_t)M * K * 2);
    __bf16* wBb = (__bf16*)carve((size_t)N * K * 2);
    __bf16* wCb = (__bf16*)carve((size_t)N * K * 2);
    __bf16* wDb = (__bf16*)carve((size_t)N * K * 2);
    float*  B0v = (float*)carve((size_t)M * N * 4);
    float*  Cv  = (float*)carve((size_t)M * N * 4);
    float*  Dv  = (float*)carve((size_t)M * N * 4);
    float*  sq  = (float*)carve((size_t)BQ * DQ * 4);
    float*  g   = (float*)carve((size_t)BQ * DQ * 4);
    float*  Pc  = (float*)carve((size_t)NC * BQ * DQ * 4);
    float*  Qc  = (float*)carve((size_t)NC * BQ * DQ * 4);

    // 1) bf16 conversions
    {
        int n = M * K;
        cvt_bf16_kernel<<<(n + 255) / 256, 256, 0, stream>>>(x, xb, n);
        n = N * K;
        cvt_bf16_kernel<<<(n + 255) / 256, 256, 0, stream>>>(W_B, wBb, n);
        cvt_bf16_kernel<<<(n + 255) / 256, 256, 0, stream>>>(W_C, wCb, n);
        cvt_bf16_kernel<<<(n + 255) / 256, 256, 0, stream>>>(W_d, wDb, n);
    }

    // 2) mean over L + SToken gate
    {
        int n = BQ * DQ;
        mean_kernel<<<(n + 255) / 256, 256, 0, stream>>>(x, sq);
        stoken_kernel<<<(n + 255) / 256, 256, 0, stream>>>(sq, st_W, st_b, st_bias, g);
    }

    // 3) three WMMA GEMMs
    {
        dim3 grid(M / BM, N / BN);   // (128, 12)
        gemm_bf16_wmma_kernel<<<grid, 256, 0, stream>>>(xb, wBb, b_B, nullptr, B0v, M, N, K, 0);
        gemm_bf16_wmma_kernel<<<grid, 256, 0, stream>>>(xb, wCb, b_C, nullptr, Cv,  M, N, K, 0);
        gemm_bf16_wmma_kernel<<<grid, 256, 0, stream>>>(xb, wDb, b_d, delta_p, Dv, M, N, K, 1);
    }

    // 4) chunked linear recurrence + gated output
    {
        int n = NC * BQ * DQ;        // 98304 threads
        scan_phase1_kernel<<<(n + 255) / 256, 256, 0, stream>>>(x, Dv, B0v, A, Pc, Qc);
        scan_phase2_kernel<<<(n + 255) / 256, 256, 0, stream>>>(x, Dv, B0v, Cv, A, g,
                                                                Pc, Qc, out);
    }
}
